// Mesh2Mesh_32091995635869
// MI455X (gfx1250) — matrix operand order repaired and checked by reading, not compiled, not run on hardware
//
#include <hip/hip_runtime.h>

typedef __attribute__((ext_vector_type(16))) __bf16 v16bf;
typedef __attribute__((ext_vector_type(8)))  float  v8f;

union AFrag { v16bf v; uint4 q[2]; };

#define DD 128

__device__ __forceinline__ unsigned short f2bfu(float f) {
    unsigned u = __builtin_bit_cast(unsigned, f);
    unsigned r = u + 0x7FFFu + ((u >> 16) & 1u);   // round-to-nearest-even
    return (unsigned short)(r >> 16);
}

__device__ __forceinline__ void st_bf4(unsigned short* p, float4 f) {
    unsigned lo = (unsigned)f2bfu(f.x) | ((unsigned)f2bfu(f.y) << 16);
    unsigned hi = (unsigned)f2bfu(f.z) | ((unsigned)f2bfu(f.w) << 16);
    ((unsigned*)p)[0] = lo;
    ((unsigned*)p)[1] = hi;
}

// ---------------------------------------------------------------------------
// Zero fill (agg buffer) — workspace is poisoned, must re-init every launch.
// ---------------------------------------------------------------------------
__global__ __launch_bounds__(256) void zero_f4(float4* __restrict__ p, int n4) {
    int i = blockIdx.x * blockDim.x + threadIdx.x;
    if (i < n4) p[i] = make_float4(0.f, 0.f, 0.f, 0.f);
}

// ---------------------------------------------------------------------------
// Pack W (Kdim x 128, row-major f32) into WMMA bf16 B-fragment layout:
// out dword index = ((k*8 + j)*32 + lane)*8 + v
//   lane<16 : col = 16j+lane,    K = 32k + 2v + (v>=4 ? 8 : 0), +1
//   lane>=16: col = 16j+lane-16, K = 32k + 8 + 2v + (v>=4 ? 8 : 0), +1
// ---------------------------------------------------------------------------
__global__ __launch_bounds__(256) void pack_w(const float* __restrict__ W,
                                              unsigned* __restrict__ out,
                                              int nchunks) {
    int idx = blockIdx.x * blockDim.x + threadIdx.x;
    if (idx >= nchunks * 2048) return;
    int v    = idx & 7;
    int lane = (idx >> 3) & 31;
    int j    = (idx >> 8) & 7;
    int k    = idx >> 11;
    int col  = 16 * j + (lane & 15);
    int kro  = 32 * k + ((lane >> 4) ? 8 : 0) + 2 * v + ((v >= 4) ? 8 : 0);
    unsigned lo = f2bfu(W[(size_t)kro * DD + col]);
    unsigned hi = f2bfu(W[(size_t)(kro + 1) * DD + col]);
    out[idx] = lo | (hi << 16);
}

// ---------------------------------------------------------------------------
// Edge kernel: delta_bond = LN(tanh([bond|src|dst] @ Wn2e)); out1 = bond + db;
// agg[src] += db  (segment sum via f32 atomics, L2-resident table)
// ---------------------------------------------------------------------------
__global__ __launch_bounds__(256) void edge_kernel(
    const float* __restrict__ bond, const float* __restrict__ node,
    const int* __restrict__ pairs, const unsigned* __restrict__ Wp,
    const float* __restrict__ g1, const float* __restrict__ b1,
    float* __restrict__ out1, float* __restrict__ agg)
{
    __shared__ __align__(16) unsigned short sA[16][392];  // 16 x 384 bf16 (+pad)
    __shared__ float sT[16][132];                         // tanh(xW), 16 x 128
    __shared__ float sP[16][16], sQ[16][16];
    __shared__ float sMean[16], sRstd[16];
    __shared__ int   sSrc[16];

    const int tid  = threadIdx.x;
    const int lane = tid & 31;
    const int wave = tid >> 5;
    const long tile = blockIdx.x;

    // ---- stage A tile: one row per wave, two passes --------------------
    #pragma unroll
    for (int half = 0; half < 2; ++half) {
        const int  row = wave + half * 8;
        const long e   = tile * 16 + row;
        const int  s   = pairs[2 * e];
        const int  d   = pairs[2 * e + 1];
        if (lane == 0) sSrc[row] = s;
        float4 fb = ((const float4*)(bond + (size_t)e * DD))[lane];
        float4 fs = ((const float4*)(node + (size_t)s * DD))[lane];
        float4 fd = ((const float4*)(node + (size_t)d * DD))[lane];
        st_bf4(&sA[row][0   + lane * 4], fb);
        st_bf4(&sA[row][128 + lane * 4], fs);
        st_bf4(&sA[row][256 + lane * 4], fd);
    }
    __syncthreads();

    // ---- GEMM: 12 chunks of 16x16x32 bf16 WMMA -------------------------
    v8f acc = {};
    const int rowA = lane & 15;
    const int kb   = (lane & 16) ? 8 : 0;  // CDNA5 A-fragment K base per half-wave
    #pragma unroll
    for (int k = 0; k < 12; ++k) {
        AFrag a, b;
        a.q[0] = *(const uint4*)&sA[rowA][32 * k + kb];
        a.q[1] = *(const uint4*)&sA[rowA][32 * k + kb + 16];
        const uint4* wp = (const uint4*)Wp + ((size_t)(k * 8 + wave) * 32 + lane) * 2;
        b.q[0] = wp[0];
        b.q[1] = wp[1];
        if (k < 11) __builtin_prefetch((const void*)(wp + 512 / 16 * 8), 0, 1);
        acc = __builtin_amdgcn_wmma_f32_16x16x32_bf16(
            false, a.v, false, b.v, (short)0, acc, false, false);
    }

    // ---- tanh, spill D tile to LDS (C/D layout: VGPR r -> M=r / r+8) ---
    const int colT = wave * 16 + (lane & 15);
    const int rb   = (lane >> 4) * 8;
    #pragma unroll
    for (int r = 0; r < 8; ++r) sT[rb + r][colT] = tanhf(acc[r]);
    __syncthreads();

    // ---- LayerNorm statistics over full 128-wide rows ------------------
    const int rrow = tid >> 4, rseg = tid & 15;
    float s1 = 0.f, s2 = 0.f;
    #pragma unroll
    for (int i = 0; i < 8; ++i) {
        float v = sT[rrow][rseg + 16 * i];
        s1 += v; s2 += v * v;
    }
    sP[rrow][rseg] = s1; sQ[rrow][rseg] = s2;
    __syncthreads();
    if (tid < 16) {
        float m = 0.f, q = 0.f;
        #pragma unroll
        for (int i = 0; i < 16; ++i) { m += sP[tid][i]; q += sQ[tid][i]; }
        m *= (1.f / 128.f);
        q  = q * (1.f / 128.f) - m * m;
        sMean[tid] = m;
        sRstd[tid] = rsqrtf(q + 1e-5f);
    }
    __syncthreads();

    // ---- apply LN, residual store, atomic scatter ----------------------
    const long e    = tile * 16 + rrow;
    const int  srcn = sSrc[rrow];
    const float m = sMean[rrow], rs = sRstd[rrow];
    #pragma unroll
    for (int i = 0; i < 8; ++i) {
        int c = rseg + 16 * i;
        float v = (sT[rrow][c] - m) * rs * g1[c] + b1[c];
        out1[(size_t)e * DD + c] = bond[(size_t)e * DD + c] + v;
        atomicAdd(&agg[(size_t)srcn * DD + c], v);
    }
}

// ---------------------------------------------------------------------------
// Node kernel: delta_node = LN(tanh([node|agg/deg] @ We2n)); out2 = node + dn
// ---------------------------------------------------------------------------
__global__ __launch_bounds__(256) void node_kernel(
    const float* __restrict__ node, const float* __restrict__ agg,
    const float* __restrict__ degrees, const unsigned* __restrict__ Wp,
    const float* __restrict__ g2, const float* __restrict__ b2,
    float* __restrict__ out2)
{
    __shared__ __align__(16) unsigned short sA[16][264];  // 16 x 256 bf16 (+pad)
    __shared__ float sT[16][132];
    __shared__ float sP[16][16], sQ[16][16];
    __shared__ float sMean[16], sRstd[16];

    const int tid  = threadIdx.x;
    const int lane = tid & 31;
    const int wave = tid >> 5;
    const long tile = blockIdx.x;

    #pragma unroll
    for (int half = 0; half < 2; ++half) {
        const int  row = wave + half * 8;
        const long n   = tile * 16 + row;
        const float inv = 1.f / degrees[n];
        float4 fn = ((const float4*)(node + (size_t)n * DD))[lane];
        float4 fa = ((const float4*)(agg  + (size_t)n * DD))[lane];
        fa.x *= inv; fa.y *= inv; fa.z *= inv; fa.w *= inv;
        st_bf4(&sA[row][0   + lane * 4], fn);
        st_bf4(&sA[row][128 + lane * 4], fa);
    }
    __syncthreads();

    v8f acc = {};
    const int rowA = lane & 15;
    const int kb   = (lane & 16) ? 8 : 0;
    #pragma unroll
    for (int k = 0; k < 8; ++k) {
        AFrag a, b;
        a.q[0] = *(const uint4*)&sA[rowA][32 * k + kb];
        a.q[1] = *(const uint4*)&sA[rowA][32 * k + kb + 16];
        const uint4* wp = (const uint4*)Wp + ((size_t)(k * 8 + wave) * 32 + lane) * 2;
        b.q[0] = wp[0];
        b.q[1] = wp[1];
        acc = __builtin_amdgcn_wmma_f32_16x16x32_bf16(
            false, a.v, false, b.v, (short)0, acc, false, false);
    }

    const int colT = wave * 16 + (lane & 15);
    const int rb   = (lane >> 4) * 8;
    #pragma unroll
    for (int r = 0; r < 8; ++r) sT[rb + r][colT] = tanhf(acc[r]);
    __syncthreads();

    const int rrow = tid >> 4, rseg = tid & 15;
    float s1 = 0.f, s2 = 0.f;
    #pragma unroll
    for (int i = 0; i < 8; ++i) {
        float v = sT[rrow][rseg + 16 * i];
        s1 += v; s2 += v * v;
    }
    sP[rrow][rseg] = s1; sQ[rrow][rseg] = s2;
    __syncthreads();
    if (tid < 16) {
        float m = 0.f, q = 0.f;
        #pragma unroll
        for (int i = 0; i < 16; ++i) { m += sP[tid][i]; q += sQ[tid][i]; }
        m *= (1.f / 128.f);
        q  = q * (1.f / 128.f) - m * m;
        sMean[tid] = m;
        sRstd[tid] = rsqrtf(q + 1e-5f);
    }
    __syncthreads();

    const long n = tile * 16 + rrow;
    const float m = sMean[rrow], rs = sRstd[rrow];
    #pragma unroll
    for (int i = 0; i < 8; ++i) {
        int c = rseg + 16 * i;
        float v = (sT[rrow][c] - m) * rs * g2[c] + b2[c];
        out2[(size_t)n * DD + c] = node[(size_t)n * DD + c] + v;
    }
}

// ---------------------------------------------------------------------------
extern "C" void kernel_launch(void* const* d_in, const int* in_sizes, int n_in,
                              void* d_out, int out_size, void* d_ws, size_t ws_size,
                              hipStream_t stream)
{
    const float* bond    = (const float*)d_in[0];  // (1,E,128)
    const float* node    = (const float*)d_in[1];  // (1,N,128)
    const int*   pairs   = (const int*)  d_in[2];  // (E,2)
    const float* degrees = (const float*)d_in[3];  // (N,1)
    const float* Wn2e    = (const float*)d_in[4];  // (384,128)
    const float* g1      = (const float*)d_in[5];
    const float* b1      = (const float*)d_in[6];
    const float* We2n    = (const float*)d_in[7];  // (256,128)
    const float* g2      = (const float*)d_in[8];
    const float* b2      = (const float*)d_in[9];

    const int E = in_sizes[2] / 2;
    const int N = in_sizes[3];

    float* out1 = (float*)d_out;
    float* out2 = out1 + (size_t)E * DD;

    char*     ws  = (char*)d_ws;
    float*    agg = (float*)ws;                             // N*128 f32
    unsigned* WpE = (unsigned*)(ws + (size_t)N * DD * 4);   // 12*2048 dwords
    unsigned* WpN = WpE + 12 * 2048;                        //  8*2048 dwords

    const int n4 = N * DD / 4;
    zero_f4 <<<(n4 + 255) / 256, 256, 0, stream>>>((float4*)agg, n4);
    pack_w  <<<(12 * 2048 + 255) / 256, 256, 0, stream>>>(Wn2e, WpE, 12);
    pack_w  <<<( 8 * 2048 + 255) / 256, 256, 0, stream>>>(We2n, WpN, 8);

    edge_kernel<<<E / 16, 256, 0, stream>>>(bond, node, pairs, WpE, g1, b1,
                                            out1, agg);
    node_kernel<<<N / 16, 256, 0, stream>>>(node, agg, degrees, WpN, g2, b2,
                                            out2);
}
